// RWKV_Tmix_x060_state_44117904065140
// MI455X (gfx1250) — matrix operand order repaired and checked by reading, not compiled, not run on hardware
//
#include <hip/hip_runtime.h>
#include <math.h>

typedef __bf16 bf16;
typedef __attribute__((ext_vector_type(16))) __bf16 bf16x16;
typedef __attribute__((ext_vector_type(8)))  __bf16 bf16x8;
typedef __attribute__((ext_vector_type(8)))  float  f32x8;
typedef __attribute__((ext_vector_type(4)))  int    i32x4;

#define Bsz   8
#define Tsz   1024
#define Csz   2048
#define Hh    32
#define Nn    64
#define Mtot  (Bsz * Tsz)          // 8192 tokens
#define DMIX5 160                  // 5 * 32
#define EPSGN 6.4e-4f              // 1e-5 * 8^2

#if __has_builtin(__builtin_amdgcn_global_load_async_to_lds_b128) && \
    __has_builtin(__builtin_amdgcn_s_wait_asynccnt)
#define HAVE_ASYNC_LDS 1
typedef __attribute__((address_space(1))) i32x4 g_i32x4;   // global int4
typedef __attribute__((address_space(3))) i32x4 l_i32x4;   // LDS int4
#else
#define HAVE_ASYNC_LDS 0
#endif

// copy 16B global -> LDS (async when available, sync load+ds_store otherwise)
__device__ __forceinline__ void cp16_g2l(const bf16* g, bf16* l) {
#if HAVE_ASYNC_LDS
  __builtin_amdgcn_global_load_async_to_lds_b128((g_i32x4*)g, (l_i32x4*)l, 0, 0);
#else
  *(bf16x8*)l = *(const bf16x8*)g;
#endif
}
__device__ __forceinline__ void cp_wait() {
#if HAVE_ASYNC_LDS
  __builtin_amdgcn_s_wait_asynccnt(0);
#endif
}

// ---------------------------------------------------------------------------
// fp32 -> bf16 bulk convert
// ---------------------------------------------------------------------------
__global__ void f32_to_bf16_kernel(const float* __restrict__ src,
                                   bf16* __restrict__ dst, int n) {
  int i = blockIdx.x * blockDim.x + threadIdx.x;
  if (i < n) dst[i] = (bf16)src[i];
}

// ---------------------------------------------------------------------------
// LoRA stage 1: lout[m, d] = tanh( sum_c xxx[m,c] * w1[c,d] ), d < 160
// ---------------------------------------------------------------------------
__global__ __launch_bounds__(160)
void lora1_kernel(const float* __restrict__ x, const float* __restrict__ maa_x,
                  const float* __restrict__ w1, float* __restrict__ lout) {
  __shared__ float sx[Csz];
  const int m = blockIdx.x;
  const int t = m & (Tsz - 1);
  const float* xr = x + (size_t)m * Csz;
  for (int c = threadIdx.x; c < Csz; c += 160) {
    float cur  = xr[c];
    float prev = (t > 0) ? xr[c - Csz] : 0.0f;
    sx[c] = cur + (prev - cur) * maa_x[c];
  }
  __syncthreads();
  const int d = threadIdx.x;            // 0..159
  float acc = 0.0f;
  for (int c = 0; c < Csz; ++c)
    acc = fmaf(sx[c], w1[c * DMIX5 + d], acc);
  lout[(size_t)m * DMIX5 + d] = tanhf(acc);
}

// ---------------------------------------------------------------------------
// 5-way mix: produce bf16 xw/xk/xv/xr/xg. One block per token.
// ---------------------------------------------------------------------------
__global__ __launch_bounds__(256)
void mix2_kernel(const float* __restrict__ x, const float* __restrict__ lora,
                 const float* __restrict__ w2,
                 const float* __restrict__ maa_w, const float* __restrict__ maa_k,
                 const float* __restrict__ maa_v, const float* __restrict__ maa_r,
                 const float* __restrict__ maa_g,
                 bf16* __restrict__ xw, bf16* __restrict__ xk,
                 bf16* __restrict__ xv, bf16* __restrict__ xr_o,
                 bf16* __restrict__ xg) {
  __shared__ float sl[DMIX5];
  const int m = blockIdx.x;
  const int t = m & (Tsz - 1);
  if (threadIdx.x < DMIX5) sl[threadIdx.x] = lora[(size_t)m * DMIX5 + threadIdx.x];
  __syncthreads();
  const float* xrow = x + (size_t)m * Csz;
  for (int c = threadIdx.x; c < Csz; c += 256) {
    float cur  = xrow[c];
    float prev = (t > 0) ? xrow[c - Csz] : 0.0f;
    float xx   = prev - cur;
    float mm[5];
#pragma unroll
    for (int f = 0; f < 5; ++f) {
      float a = 0.0f;
#pragma unroll
      for (int d = 0; d < 32; ++d)
        a = fmaf(sl[f * 32 + d], w2[(size_t)(f * 32 + d) * Csz + c], a);
      mm[f] = a;
    }
    size_t idx = (size_t)m * Csz + c;
    xw[idx]   = (bf16)(cur + xx * (maa_w[c] + mm[0]));
    xk[idx]   = (bf16)(cur + xx * (maa_k[c] + mm[1]));
    xv[idx]   = (bf16)(cur + xx * (maa_v[c] + mm[2]));
    xr_o[idx] = (bf16)(cur + xx * (maa_r[c] + mm[3]));
    xg[idx]   = (bf16)(cur + xx * (maa_g[c] + mm[4]));
  }
}

// ---------------------------------------------------------------------------
// WMMA GEMM, LDS-staged + double-buffered async copies.
//   D[m,n] = sum_k X[m,k] * W[n,k]   (X: MxK bf16, W: NxK bf16)
// Block = 256 threads = 8 waves (4 M x 2 N) -> 128x128 tile; wave = 32x64
// (2x4 fragments, 8 f32x8 accumulators). K staged in 64-wide slices through
// LDS (rows padded to 72 elems = 144B to spread banks); fragments read with
// ds_load_b128; next slice streams in via GLOBAL_LOAD_ASYNC_TO_LDS while the
// current slice's 16 v_wmma execute.
// Fragment layouts per CDNA5 ISA 7.12.2 (A: lane=M, hi half picks K-range;
// B: lane=N, hi picks K 8..15 / 24..31; C/D: vgpr v -> M = v + 8*hi, N = lane&15).
// op: 0 = none, 1 = SiLU
// ---------------------------------------------------------------------------
#define RS 72                 // padded LDS row stride (elements)
#define TS (128 * RS)         // one staged tile

union ABfrag { bf16x16 v; bf16x8 h[2]; };

__global__ __launch_bounds__(256)
void gemm_xwt_bf16_kernel(const bf16* __restrict__ X, const bf16* __restrict__ W,
                          float* __restrict__ D, int K, int ldD, int op) {
  __shared__ bf16 lA[2 * TS];
  __shared__ bf16 lB[2 * TS];

  const int tid  = threadIdx.x;
  const int lane = tid & 31;
  const int wave = tid >> 5;
  const int wr   = wave & 3;     // 4 wave rows (M)
  const int wc   = wave >> 2;    // 2 wave cols (N)
  const int lm   = lane & 15;
  const int hi   = lane >> 4;

  const int mBlk = blockIdx.y * 128;
  const int nBlk = blockIdx.x * 128;

  // staging map: thread t copies half a row (32 elems = 4 x 16B) of A and B
  const int srow = tid >> 1;            // 0..127
  const int scol = (tid & 1) * 32;      // 0 or 32
  const bf16* gA = X + (size_t)(mBlk + srow) * K + scol;
  const bf16* gB = W + (size_t)(nBlk + srow) * K + scol;
  bf16* sA = &lA[srow * RS + scol];
  bf16* sB = &lB[srow * RS + scol];

  auto stage = [&](int buf, int k0) {
#pragma unroll
    for (int i = 0; i < 4; ++i) {
      cp16_g2l(gA + k0 + 8 * i, sA + buf * TS + 8 * i);
      cp16_g2l(gB + k0 + 8 * i, sB + buf * TS + 8 * i);
    }
  };

  f32x8 acc[2][4];
#pragma unroll
  for (int mi = 0; mi < 2; ++mi)
#pragma unroll
    for (int ni = 0; ni < 4; ++ni) acc[mi][ni] = (f32x8){};

  const bf16* fA0 = &lA[(wr * 32 + lm) * RS];
  const bf16* fA1 = &lA[(wr * 32 + 16 + lm) * RS];
  const bf16* fB[4];
#pragma unroll
  for (int ni = 0; ni < 4; ++ni)
    fB[ni] = &lB[(wc * 64 + 16 * ni + lm) * RS];

  stage(0, 0);
  const int nst = K / 64;
  for (int s = 0; s < nst; ++s) {
    const int cur = s & 1;
    cp_wait();                 // my async copies for buf `cur` are done
    __syncthreads();           // everyone's copies are visible
    if (s + 1 < nst) stage(cur ^ 1, (s + 1) * 64);
    if (s + 2 < nst) {         // keep HBM/L2 ahead of the pipeline
      __builtin_prefetch(gA + (s + 2) * 64, 0, 3);
      __builtin_prefetch(gB + (s + 2) * 64, 0, 3);
    }
    const int bo = cur * TS;
#pragma unroll
    for (int ks = 0; ks < 64; ks += 32) {
      ABfrag a0, a1;
      a0.h[0] = *(const bf16x8*)(fA0 + bo + ks + 8 * hi);
      a0.h[1] = *(const bf16x8*)(fA0 + bo + ks + 16 + 8 * hi);
      a1.h[0] = *(const bf16x8*)(fA1 + bo + ks + 8 * hi);
      a1.h[1] = *(const bf16x8*)(fA1 + bo + ks + 16 + 8 * hi);
#pragma unroll
      for (int ni = 0; ni < 4; ++ni) {
        ABfrag b;
        b.h[0] = *(const bf16x8*)(fB[ni] + bo + ks + 8 * hi);
        b.h[1] = *(const bf16x8*)(fB[ni] + bo + ks + 16 + 8 * hi);
        acc[0][ni] = __builtin_amdgcn_wmma_f32_16x16x32_bf16(
            false, a0.v, false, b.v, (short)0, acc[0][ni], false, false);
        acc[1][ni] = __builtin_amdgcn_wmma_f32_16x16x32_bf16(
            false, a1.v, false, b.v, (short)0, acc[1][ni], false, false);
      }
    }
    __syncthreads();           // all reads of buf `cur` done before re-stage
  }

#pragma unroll
  for (int mi = 0; mi < 2; ++mi) {
#pragma unroll
    for (int ni = 0; ni < 4; ++ni) {
#pragma unroll
      for (int v = 0; v < 8; ++v) {
        int mr = mBlk + wr * 32 + 16 * mi + 8 * hi + v;
        int nc = nBlk + wc * 64 + 16 * ni + lm;
        float d = acc[mi][ni][v];
        if (op == 1) d = d / (1.0f + __expf(-d));
        D[(size_t)mr * ldD + nc] = d;
      }
    }
  }
}

// ---------------------------------------------------------------------------
// Decay: w = time_decay + tanh(xw @ dw1) @ dw2 ; decay = exp(-exp(w))
// ---------------------------------------------------------------------------
__global__ __launch_bounds__(256)
void decay_kernel(const bf16* __restrict__ xw, const float* __restrict__ tdec,
                  const float* __restrict__ dw1, const float* __restrict__ dw2,
                  float* __restrict__ decay) {
  __shared__ float sx[Csz];
  __shared__ float part[256];
  __shared__ float t1[64];
  const int m = blockIdx.x;
  for (int c = threadIdx.x; c < Csz; c += 256)
    sx[c] = (float)xw[(size_t)m * Csz + c];
  __syncthreads();
  const int j   = threadIdx.x & 63;
  const int seg = threadIdx.x >> 6;     // 4 segments of 512
  float acc = 0.0f;
  const int c0 = seg * 512;
  for (int c = c0; c < c0 + 512; ++c)
    acc = fmaf(sx[c], dw1[c * 64 + j], acc);
  part[threadIdx.x] = acc;
  __syncthreads();
  if (threadIdx.x < 64) {
    float s = part[threadIdx.x] + part[64 + threadIdx.x] +
              part[128 + threadIdx.x] + part[192 + threadIdx.x];
    t1[threadIdx.x] = tanhf(s);
  }
  __syncthreads();
  for (int c = threadIdx.x; c < Csz; c += 256) {
    float a = tdec[c];
#pragma unroll 8
    for (int jj = 0; jj < 64; ++jj)
      a = fmaf(t1[jj], dw2[jj * Csz + c], a);
    decay[(size_t)m * Csz + c] = __expf(-__expf(a));
  }
}

// ---------------------------------------------------------------------------
// WKV6 recurrence. One block (64 threads) per (b,h); thread i owns column i
// of the 64x64 state (S[j][i] in 64 VGPRs). Serial over T.
// ---------------------------------------------------------------------------
__global__ __launch_bounds__(64)
void wkv_kernel(const float* __restrict__ r, const float* __restrict__ k,
                const float* __restrict__ v, const float* __restrict__ dcy,
                const float* __restrict__ u_, const float* __restrict__ s0,
                float* __restrict__ y) {
  const int h = blockIdx.x & (Hh - 1);
  const int b = blockIdx.x >> 5;
  const int i = threadIdx.x;            // 0..63 (value dim)
  __shared__ float sr[64], sk[64], sd[64], su[64];
  su[i] = u_[h * 64 + i];
  float S[64];
#pragma unroll
  for (int j = 0; j < 64; ++j)
    S[j] = s0[((size_t)h * 64 + j) * 64 + i];
  __syncthreads();
  size_t base = (size_t)b * Tsz * Csz + (size_t)h * 64;
  for (int t = 0; t < Tsz; ++t, base += Csz) {
    sr[i] = r[base + i];
    sk[i] = k[base + i];
    sd[i] = dcy[base + i];
    float vi = v[base + i];
    __syncthreads();
    float acc = 0.0f;
#pragma unroll
    for (int j = 0; j < 64; ++j) {
      float kv = sk[j] * vi;
      acc  = fmaf(sr[j], fmaf(su[j], kv, S[j]), acc);
      S[j] = fmaf(sd[j], S[j], kv);
    }
    y[base + i] = acc;
    __syncthreads();
  }
}

// ---------------------------------------------------------------------------
// GroupNorm over heads + multiply by g, emit bf16 activation for W_o GEMM.
// ---------------------------------------------------------------------------
__global__ __launch_bounds__(256)
void gnorm_kernel(const float* __restrict__ y, const float* __restrict__ g,
                  const float* __restrict__ lnw, const float* __restrict__ lnb,
                  bf16* __restrict__ act) {
  __shared__ float ps[256], pq[256];
  const int m   = blockIdx.x;
  const int tid = threadIdx.x;
  const size_t base = (size_t)m * Csz;
  float vv[8];
  float s = 0.0f, q = 0.0f;
#pragma unroll
  for (int e = 0; e < 8; ++e) {
    float a = y[base + tid * 8 + e];
    vv[e] = a; s += a; q += a * a;
  }
  ps[tid] = s; pq[tid] = q;
  __syncthreads();
  const int h8 = (tid >> 3) << 3;       // first of the 8 threads in my head
  float S = 0.0f, Q = 0.0f;
#pragma unroll
  for (int i = 0; i < 8; ++i) { S += ps[h8 + i]; Q += pq[h8 + i]; }
  float mu   = S * (1.0f / 64.0f);
  float var  = Q * (1.0f / 64.0f) - mu * mu;
  float rstd = rsqrtf(var + EPSGN);
#pragma unroll
  for (int e = 0; e < 8; ++e) {
    int c = tid * 8 + e;
    float yn = (vv[e] - mu) * rstd * lnw[c] + lnb[c];
    act[base + c] = (bf16)(yn * g[base + c]);
  }
}

// ---------------------------------------------------------------------------
// e1/e2 bookkeeping from last-timestep k.
// ---------------------------------------------------------------------------
__global__ void e1e2_kernel(const float* __restrict__ k, const float* __restrict__ u,
                            float* __restrict__ e1, float* __restrict__ e2) {
  int idx = blockIdx.x * blockDim.x + threadIdx.x;
  if (idx >= Bsz * Csz) return;
  int b = idx / Csz, c = idx % Csz;
  float kl = k[((size_t)b * Tsz + (Tsz - 1)) * Csz + c];
  float uu = u[c];
  float p  = fmaxf(uu, kl);
  e1[idx] = __expf(uu - p);
  e2[idx] = __expf(kl - p);
}

// ---------------------------------------------------------------------------
extern "C" void kernel_launch(void* const* d_in, const int* in_sizes, int n_in,
                              void* d_out, int out_size, void* d_ws, size_t ws_size,
                              hipStream_t stream) {
  const float* x     = (const float*)d_in[0];
  const float* maa_x = (const float*)d_in[1];
  const float* maa_w = (const float*)d_in[2];
  const float* maa_k = (const float*)d_in[3];
  const float* maa_v = (const float*)d_in[4];
  const float* maa_r = (const float*)d_in[5];
  const float* maa_g = (const float*)d_in[6];
  const float* w1    = (const float*)d_in[7];
  const float* w2    = (const float*)d_in[8];
  const float* tdec  = (const float*)d_in[9];
  const float* dw1   = (const float*)d_in[10];
  const float* dw2   = (const float*)d_in[11];
  const float* u     = (const float*)d_in[12];
  const float* s0    = (const float*)d_in[13];
  const float* Wf[5] = { (const float*)d_in[14], (const float*)d_in[15],
                         (const float*)d_in[16], (const float*)d_in[17],
                         (const float*)d_in[18] };        // r,k,v,g,o
  const float* lnw   = (const float*)d_in[19];
  const float* lnb   = (const float*)d_in[20];

  float* out = (float*)d_out;
  float* e1  = out + (size_t)Mtot * Csz;
  float* e2  = e1 + (size_t)Bsz * Csz;

  // ---- workspace carve-up (256B aligned) ----
  char* ws = (char*)d_ws;
  size_t off = 0;
  auto carve = [&](size_t bytes) -> char* {
    off = (off + 255) & ~(size_t)255;
    char* p = ws + off;
    off += bytes;
    return p;
  };
  const size_t WWb  = (size_t)Csz * Csz * sizeof(bf16);   // 8.4 MB
  const size_t ACTb = (size_t)Mtot * Csz * sizeof(bf16);  // 33.6 MB
  const size_t ACTf = (size_t)Mtot * Csz * sizeof(float); // 67.1 MB

  bf16* Wbf[5];
  for (int i = 0; i < 5; ++i) Wbf[i] = (bf16*)carve(WWb);
  bf16* xw_bf = (bf16*)carve(ACTb);
  bf16* xk_bf = (bf16*)carve(ACTb);
  bf16* xv_bf = (bf16*)carve(ACTb);
  bf16* xr_bf = (bf16*)carve(ACTb);
  bf16* xg_bf = (bf16*)carve(ACTb);
  float* lora = (float*)carve((size_t)Mtot * DMIX5 * sizeof(float));
  float* rbuf = (float*)carve(ACTf);
  float* kbuf = (float*)carve(ACTf);
  float* vbuf = (float*)carve(ACTf);
  float* dbuf = (float*)carve(ACTf);
  float* gbuf = (float*)carve(ACTf);
  float* ybuf = (float*)carve(ACTf);
  bf16*  acto = (bf16*)carve(ACTb);
  (void)ws_size; (void)in_sizes; (void)n_in; (void)out_size;

  // 1) weights -> bf16
  {
    int n = Csz * Csz;
    dim3 g((n + 255) / 256);
    for (int i = 0; i < 5; ++i)
      f32_to_bf16_kernel<<<g, 256, 0, stream>>>(Wf[i], Wbf[i], n);
  }
  // 2) LoRA stage 1
  lora1_kernel<<<Mtot, 160, 0, stream>>>(x, maa_x, w1, lora);
  // 3) 5-way mix -> bf16 activations
  mix2_kernel<<<Mtot, 256, 0, stream>>>(x, lora, w2, maa_w, maa_k, maa_v,
                                        maa_r, maa_g, xw_bf, xk_bf, xv_bf,
                                        xr_bf, xg_bf);
  // 4) big WMMA GEMMs: r, k, v, g(SiLU).  128x128 block tile.
  dim3 gg(Csz / 128, Mtot / 128);
  gemm_xwt_bf16_kernel<<<gg, 256, 0, stream>>>(xr_bf, Wbf[0], rbuf, Csz, Csz, 0);
  gemm_xwt_bf16_kernel<<<gg, 256, 0, stream>>>(xk_bf, Wbf[1], kbuf, Csz, Csz, 0);
  gemm_xwt_bf16_kernel<<<gg, 256, 0, stream>>>(xv_bf, Wbf[2], vbuf, Csz, Csz, 0);
  gemm_xwt_bf16_kernel<<<gg, 256, 0, stream>>>(xg_bf, Wbf[3], gbuf, Csz, Csz, 1);
  // 5) e1/e2 (needs only k)
  e1e2_kernel<<<(Bsz * Csz + 255) / 256, 256, 0, stream>>>(kbuf, u, e1, e2);
  // 6) decay
  decay_kernel<<<Mtot, 256, 0, stream>>>(xw_bf, tdec, dw1, dw2, dbuf);
  // 7) WKV6 scan
  wkv_kernel<<<Bsz * Hh, 64, 0, stream>>>(rbuf, kbuf, vbuf, dbuf, u, s0, ybuf);
  // 8) GroupNorm * g -> bf16
  gnorm_kernel<<<Mtot, 256, 0, stream>>>(ybuf, gbuf, lnw, lnb, acto);
  // 9) output WMMA GEMM
  gemm_xwt_bf16_kernel<<<gg, 256, 0, stream>>>(acto, Wbf[4], out, Csz, Csz, 0);
}
